// CustomTransformerModel_50964081934666
// MI455X (gfx1250) — matrix operand order repaired and checked by reading, compiled, not run on hardware
//
#include <hip/hip_runtime.h>
#include <hip/hip_bf16.h>

// ---------------------------------------------------------------------------
// Model dims (from reference)
// ---------------------------------------------------------------------------
static constexpr int Ll   = 4;
static constexpr int Dd   = 1024;
static constexpr int Hh   = 16;
static constexpr int HDim = 64;        // D / H
static constexpr int Ff   = 4096;
static constexpr int Vv   = 32000;
static constexpr int Bb   = 2;
static constexpr int Ss   = 2048;
static constexpr int Mtok = Bb * Ss;   // 4096 tokens
#define LN_EPS 1e-5f

// ---------------------------------------------------------------------------
// WMMA plumbing (CDNA5 wave32, v_wmma_f32_16x16x32_bf16)
// ---------------------------------------------------------------------------
typedef __attribute__((ext_vector_type(16))) __bf16 v16bf;
typedef __attribute__((ext_vector_type(8)))  float  v8f;
typedef __attribute__((ext_vector_type(4)))  unsigned u32x4;
typedef __attribute__((ext_vector_type(8)))  int      i32x8;
typedef __attribute__((ext_vector_type(4)))  int      i32x4;

union BF16Frag { v16bf v; unsigned u[8]; };

__device__ __forceinline__ v8f wmma_bf16(v16bf a, v16bf b, v8f c) {
  return __builtin_amdgcn_wmma_f32_16x16x32_bf16(false, a, false, b,
                                                 (short)0, c, false, false);
}

// A fragment: 16x32 bf16, row-major LDS source (K contiguous).
__device__ __forceinline__ v16bf load_a_frag(const __bf16* lds, int stride,
                                             int row0, int kbase) {
  const int lane = threadIdx.x & 31;
  const int hl   = lane >> 4;
  const int r    = lane & 15;
  const __bf16* row = lds + (size_t)(row0 + r) * stride + kbase;
  BF16Frag f;
#pragma unroll
  for (int v = 0; v < 8; ++v)
    f.u[v] = *(const unsigned*)(row + hl * 16 + 2 * v);
  return f.v;
}

// B fragment from an LDS tile stored N-contiguous per K row  (sB[k][n], pitch
// elems). Lane=col, K-pairs packed from two 16-bit reads.
__device__ __forceinline__ v16bf load_b_frag_rm(const __bf16* sB, int pitch,
                                                int n0, int kbase) {
  const int lane = threadIdx.x & 31;
  const int hl   = lane >> 4;
  const int c    = lane & 15;
  const unsigned short* p = (const unsigned short*)sB;
  BF16Frag f;
#pragma unroll
  for (int v = 0; v < 8; ++v) {
    const int k = kbase + hl * 16 + 2 * v;
    const unsigned lo = p[(size_t)k * pitch + n0 + c];
    const unsigned hi = p[(size_t)(k + 1) * pitch + n0 + c];
    f.u[v] = lo | (hi << 16);
  }
  return f.v;
}

// B fragment from K-contiguous transposed LDS (ldsT[n][k]).
__device__ __forceinline__ v16bf load_b_frag(const __bf16* ldsT, int stride,
                                             int n0, int kbase) {
  const int lane = threadIdx.x & 31;
  const int hl   = lane >> 4;
  const int c    = lane & 15;
  const __bf16* row = ldsT + (size_t)(n0 + c) * stride + kbase + hl * 16;
  BF16Frag f;
#pragma unroll
  for (int v = 0; v < 8; ++v)
    f.u[v] = *(const unsigned*)(row + 2 * v);
  return f.v;
}

// ---------------------------------------------------------------------------
// CDNA5 async-to-LDS + Tensor Data Mover helpers
// ---------------------------------------------------------------------------
__device__ __forceinline__ void async_load_b128(void* lds_dst, const void* gsrc) {
  const unsigned loff = (unsigned)(size_t)lds_dst;   // flat addr[31:0] == LDS offset
  asm volatile("global_load_async_to_lds_b128 %0, %1, off"
               :: "v"(loff), "v"(gsrc) : "memory");
}
__device__ __forceinline__ void wait_async0() {
  asm volatile("s_wait_asynccnt 0" ::: "memory");
}

// 2D TDM tile load: tensor of `width` x `rows` bf16 elems with row stride
// `stride` elems; copies tile_w x tile_h (rows packed back-to-back in LDS).
__device__ __forceinline__ void tdm_load_2d(void* lds_dst, const void* gsrc,
                                            unsigned width, unsigned rows,
                                            unsigned long long stride_elems,
                                            unsigned tile_w, unsigned tile_h) {
  const unsigned long long ga = (unsigned long long)(size_t)gsrc;
  u32x4 g0;
  g0[0] = 1u;                                              // count=1, user D#
  g0[1] = (unsigned)(size_t)lds_dst;                       // lds_addr
  g0[2] = (unsigned)(ga & 0xffffffffull);                  // global_addr lo
  g0[3] = (unsigned)((ga >> 32) & 0x1ffffffull) | (2u << 30); // addr hi | type=2
  i32x8 g1;
  g1[0] = (int)(1u << 16);                                 // data_size=1 (2B)
  g1[1] = (int)((width & 0xffffu) << 16);                  // tensor_dim0 lo16
  g1[2] = (int)((width >> 16) | ((rows & 0xffffu) << 16)); // dim0 hi | dim1 lo
  g1[3] = (int)((rows >> 16) | (tile_w << 16));            // dim1 hi | tile_dim0
  g1[4] = (int)(tile_h & 0xffffu);                         // tile_dim1 (dim2=0)
  g1[5] = (int)(unsigned)(stride_elems & 0xffffffffull);   // dim0_stride lo
  g1[6] = (int)(unsigned)((stride_elems >> 32) & 0xffffull); // dim0_stride hi
  g1[7] = 0;
  const i32x4 g2 = {0, 0, 0, 0};
  const i32x4 g3 = {0, 0, 0, 0};
  asm volatile("tensor_load_to_lds %0, %1, %2, %3"
               :: "s"(g0), "s"(g1), "s"(g2), "s"(g3) : "memory");
}
__device__ __forceinline__ void wait_tensor0() {
  __builtin_amdgcn_s_wait_tensorcnt(0);
}

// ---------------------------------------------------------------------------
// f32 -> bf16 conversion (weights, once per launch)
// ---------------------------------------------------------------------------
__global__ void cvt_bf16_kernel(const float* __restrict__ x,
                                __bf16* __restrict__ y, size_t n) {
  size_t i = (size_t)blockIdx.x * 256 + threadIdx.x;
  if (i < n) y[i] = (__bf16)x[i];
}

// ---------------------------------------------------------------------------
// Embedding: h = 2*tok + pe  (fp32 + bf16 copies)
// ---------------------------------------------------------------------------
__global__ void embed_kernel(const int* __restrict__ ids,
                             const float* __restrict__ emb,
                             const float* __restrict__ pe,
                             float* __restrict__ hf, __bf16* __restrict__ hb) {
  const int t = blockIdx.x;
  const int d = blockIdx.y * 256 + threadIdx.x;
  const int s = t % Ss;
  const int id = ids[t];
  float v = 2.0f * emb[(size_t)id * Dd + d] + pe[(size_t)s * Dd + d];
  hf[(size_t)t * Dd + d] = v;
  hb[(size_t)t * Dd + d] = (__bf16)v;
}

// ---------------------------------------------------------------------------
// Double-buffered bf16 WMMA GEMM: C[M,N] = A[M,K] @ B[K,N] (+bias)(+GELU)
// A tile -> async_load_to_lds, B tile -> TDM tensor_load_to_lds (wave 0).
// Block tile 128x128, K-step 32, 8 waves (2x4), each wave 64x32.
// ---------------------------------------------------------------------------
template <bool BIAS, bool GELU, bool OF32, bool OBF16>
__global__ __launch_bounds__(256)
void gemm_bf16_kernel(const __bf16* __restrict__ A, const __bf16* __restrict__ Bm,
                      const float* __restrict__ bias,
                      float* __restrict__ Cf, __bf16* __restrict__ Ch,
                      int M, int N, int K) {
  constexpr int TM = 128, TN = 128, TK = 32;
  constexpr int APAD = 48;        // A row stride (elems): 96B, 16B aligned
  constexpr int BPITCH = 128;     // B row stride (elems): TDM packs rows tight
  __shared__ __bf16 sA[2][TM * APAD];
  __shared__ __bf16 sB[2][TK * BPITCH];

  const int tid  = threadIdx.x;
  const int lane = tid & 31;
  const int wid  = tid >> 5;
  const int wm   = wid >> 2;
  const int wn   = wid & 3;
  const int bm   = blockIdx.y * TM;
  const int bn   = blockIdx.x * TN;

  const int arow = tid >> 1, ahf = tid & 1;

  auto issue_tile = [&](int k0, int buf) {
    // A tile: 128x32 bf16 via per-lane async b128 (2 instr / wave)
    const __bf16* ga = A + (size_t)(bm + arow) * K + k0 + ahf * 16;
    __bf16* la = &sA[buf][arow * APAD + ahf * 16];
    async_load_b128(la, ga);
    async_load_b128(la + 8, ga + 8);
    // B tile: 32x128 bf16 via TDM (one descriptor, wave 0 only)
    if (wid == 0) {
      tdm_load_2d(&sB[buf][0], Bm + (size_t)k0 * N + bn,
                  (unsigned)N, (unsigned)K, (unsigned long long)N,
                  (unsigned)TN, (unsigned)TK);
    }
  };

  v8f acc[4][2] = {};

  issue_tile(0, 0);
  wait_async0();
  wait_tensor0();
  __syncthreads();

  const int ksteps = K / TK;
  for (int ks = 0; ks < ksteps; ++ks) {
    const int buf = ks & 1;
    if (ks + 1 < ksteps) issue_tile((ks + 1) * TK, buf ^ 1);

    v16bf af[4], bf[2];
#pragma unroll
    for (int mt = 0; mt < 4; ++mt)
      af[mt] = load_a_frag(&sA[buf][0], APAD, wm * 64 + mt * 16, 0);
#pragma unroll
    for (int nt = 0; nt < 2; ++nt)
      bf[nt] = load_b_frag_rm(&sB[buf][0], BPITCH, wn * 32 + nt * 16, 0);
#pragma unroll
    for (int mt = 0; mt < 4; ++mt)
#pragma unroll
      for (int nt = 0; nt < 2; ++nt)
        acc[mt][nt] = wmma_bf16(af[mt], bf[nt], acc[mt][nt]);

    wait_async0();
    wait_tensor0();
    __syncthreads();
  }

  // Epilogue: lanes 0-15 N=lane M=v, lanes 16-31 N=lane-16 M=8+v
  const int hl = lane >> 4, c = lane & 15;
#pragma unroll
  for (int mt = 0; mt < 4; ++mt) {
#pragma unroll
    for (int nt = 0; nt < 2; ++nt) {
      const int col = bn + wn * 32 + nt * 16 + c;
      const float bv = BIAS ? bias[col] : 0.0f;
#pragma unroll
      for (int v = 0; v < 8; ++v) {
        const int row = bm + wm * 64 + mt * 16 + hl * 8 + v;
        float x = acc[mt][nt][v] + bv;
        if (GELU) x = 0.5f * x * (1.0f + erff(x * 0.70710678118f));
        const size_t idx = (size_t)row * N + col;
        if (OF32)  Cf[idx] = x;
        if (OBF16) Ch[idx] = (__bf16)x;
      }
    }
  }
}

// ---------------------------------------------------------------------------
// Flash-style attention. Block = 4 waves = 64 query rows of one (b,h).
// Q/K tiles via async-to-LDS; QK^T and P*V via bf16 WMMA; fp32 online softmax.
// ---------------------------------------------------------------------------
__global__ __launch_bounds__(128)
void attention_kernel(const __bf16* __restrict__ Q, const __bf16* __restrict__ Kc,
                      const __bf16* __restrict__ Vc, __bf16* __restrict__ O) {
  constexpr int PAD = 80;                 // 160B rows, 16B aligned
  __shared__ __bf16 sQ [64 * PAD];        // [q][hd]
  __shared__ __bf16 sK [64 * PAD];        // [key][hd]  (B^T for QK^T)
  __shared__ __bf16 sVt[64 * PAD];        // [hd][key]  (B^T for P*V)
  __shared__ __bf16 sP [64 * PAD];        // [q][key]

  const int tid  = threadIdx.x;
  const int wid  = tid >> 5;
  const int lane = tid & 31;
  const int hl   = lane >> 4, c = lane & 15;

  const int bh = blockIdx.y;
  const int b  = bh / Hh, h = bh % Hh;
  const int q0 = blockIdx.x * 64;

  const int row = tid >> 1, hf = tid & 1;

  // Q tile 64x64 via async copy
  {
    const __bf16* g = Q + ((size_t)(b * Ss + q0 + row)) * Dd + h * HDim + hf * 32;
    __bf16* s = sQ + row * PAD + hf * 32;
#pragma unroll
    for (int i = 0; i < 4; ++i) async_load_b128(s + i * 8, g + i * 8);
  }
  wait_async0();
  __syncthreads();

  v16bf aq[2];
  aq[0] = load_a_frag(sQ, PAD, wid * 16, 0);
  aq[1] = load_a_frag(sQ, PAD, wid * 16, 32);

  float mrow[8], lrow[8];
#pragma unroll
  for (int v = 0; v < 8; ++v) { mrow[v] = -3.0e38f; lrow[v] = 0.0f; }
  v8f accO[4] = {};

  for (int kc = 0; kc < Ss; kc += 64) {
    // K chunk [64 keys][64 hd] via async copy
    {
      const __bf16* g = Kc + ((size_t)(b * Ss + kc + row)) * Dd + h * HDim + hf * 32;
      __bf16* s = sK + row * PAD + hf * 32;
#pragma unroll
      for (int i = 0; i < 4; ++i) async_load_b128(s + i * 8, g + i * 8);
    }
    // V chunk transposed -> sVt[hd][key] (scalar; needs element transpose)
    {
      const int key = tid & 63, hg = (tid >> 6) * 32;
      const __bf16* g = Vc + ((size_t)(b * Ss + kc + key)) * Dd + h * HDim + hg;
#pragma unroll
      for (int i = 0; i < 32; ++i) sVt[(hg + i) * PAD + key] = g[i];
    }
    wait_async0();
    __syncthreads();

    // Scores (4 tiles of 16 keys, K-dim = hd = 64)
    v8f st[4];
#pragma unroll
    for (int nt = 0; nt < 4; ++nt) {
      v8f s0 = {};
      s0 = wmma_bf16(aq[0], load_b_frag(sK, PAD, nt * 16, 0),  s0);
      s0 = wmma_bf16(aq[1], load_b_frag(sK, PAD, nt * 16, 32), s0);
      st[nt] = s0;
    }

    float tmax[8];
#pragma unroll
    for (int v = 0; v < 8; ++v) tmax[v] = -3.0e38f;
#pragma unroll
    for (int nt = 0; nt < 4; ++nt)
#pragma unroll
      for (int v = 0; v < 8; ++v) {
        float x = st[nt][v] * 0.125f;
        st[nt][v] = x;
        tmax[v] = fmaxf(tmax[v], x);
      }
#pragma unroll
    for (int v = 0; v < 8; ++v) {
      float x = tmax[v];
#pragma unroll
      for (int m = 1; m < 16; m <<= 1) x = fmaxf(x, __shfl_xor(x, m, 32));
      tmax[v] = x;
    }
#pragma unroll
    for (int v = 0; v < 8; ++v) {
      const float mnew = fmaxf(mrow[v], tmax[v]);
      const float corr = expf(mrow[v] - mnew);
#pragma unroll
      for (int nt = 0; nt < 4; ++nt) accO[nt][v] *= corr;
      float rsum = 0.0f;
#pragma unroll
      for (int nt = 0; nt < 4; ++nt) {
        const float p = expf(st[nt][v] - mnew);
        st[nt][v] = p;
        rsum += p;
      }
#pragma unroll
      for (int m = 1; m < 16; m <<= 1) rsum += __shfl_xor(rsum, m, 32);
      lrow[v] = lrow[v] * corr + rsum;
      mrow[v] = mnew;
    }
    // P (C-layout) -> LDS row-major for A-fragment reload
#pragma unroll
    for (int nt = 0; nt < 4; ++nt)
#pragma unroll
      for (int v = 0; v < 8; ++v)
        sP[(wid * 16 + hl * 8 + v) * PAD + nt * 16 + c] = (__bf16)st[nt][v];
    __syncthreads();

    // accO += P @ V
#pragma unroll
    for (int ks = 0; ks < 2; ++ks) {
      v16bf ap = load_a_frag(sP, PAD, wid * 16, ks * 32);
#pragma unroll
      for (int nt = 0; nt < 4; ++nt)
        accO[nt] = wmma_bf16(ap, load_b_frag(sVt, PAD, nt * 16, ks * 32), accO[nt]);
    }
    __syncthreads();
  }

#pragma unroll
  for (int nt = 0; nt < 4; ++nt)
#pragma unroll
    for (int v = 0; v < 8; ++v) {
      const int r = q0 + wid * 16 + hl * 8 + v;
      const int col = h * HDim + nt * 16 + c;
      O[((size_t)(b * Ss + r)) * Dd + col] = (__bf16)(accO[nt][v] / lrow[v]);
    }
}

// ---------------------------------------------------------------------------
// LayerNorm (+optional residual), fp32 math, fp32+bf16 outputs. 1 token/block.
// ---------------------------------------------------------------------------
template <bool RES>
__global__ __launch_bounds__(256)
void layernorm_kernel(const float* __restrict__ X, const float* __restrict__ R,
                      const float* __restrict__ g, const float* __restrict__ bta,
                      float* __restrict__ Yf, __bf16* __restrict__ Yh) {
  __shared__ float red[256];
  const int t = blockIdx.x;
  const int tid = threadIdx.x;
  float x[4];
#pragma unroll
  for (int i = 0; i < 4; ++i) {
    const int d = tid + i * 256;
    float v = X[(size_t)t * Dd + d];
    if (RES) v += R[(size_t)t * Dd + d];
    x[i] = v;
  }
  red[tid] = x[0] + x[1] + x[2] + x[3];
  __syncthreads();
  for (int o = 128; o > 0; o >>= 1) {
    if (tid < o) red[tid] += red[tid + o];
    __syncthreads();
  }
  const float mean = red[0] * (1.0f / Dd);
  __syncthreads();
  float vs = 0.0f;
#pragma unroll
  for (int i = 0; i < 4; ++i) { const float d = x[i] - mean; vs += d * d; }
  red[tid] = vs;
  __syncthreads();
  for (int o = 128; o > 0; o >>= 1) {
    if (tid < o) red[tid] += red[tid + o];
    __syncthreads();
  }
  const float rstd = rsqrtf(red[0] * (1.0f / Dd) + LN_EPS);
#pragma unroll
  for (int i = 0; i < 4; ++i) {
    const int d = tid + i * 256;
    const float y = (x[i] - mean) * rstd * g[d] + bta[d];
    Yf[(size_t)t * Dd + d] = y;
    Yh[(size_t)t * Dd + d] = (__bf16)y;
  }
}

// ---------------------------------------------------------------------------
// Host launcher
// ---------------------------------------------------------------------------
extern "C" void kernel_launch(void* const* d_in, const int* in_sizes, int n_in,
                              void* d_out, int out_size, void* d_ws, size_t ws_size,
                              hipStream_t stream) {
  (void)in_sizes; (void)n_in; (void)out_size; (void)ws_size;

  const int*   ids    = (const int*)  d_in[0];
  const float* embw   = (const float*)d_in[1];
  const float* pe     = (const float*)d_in[2];
  const float* wq     = (const float*)d_in[3];
  const float* bq     = (const float*)d_in[4];
  const float* wk     = (const float*)d_in[5];
  const float* bk     = (const float*)d_in[6];
  const float* wv     = (const float*)d_in[7];
  const float* bv     = (const float*)d_in[8];
  const float* wo     = (const float*)d_in[9];
  const float* bo     = (const float*)d_in[10];
  const float* ln1g   = (const float*)d_in[11];
  const float* ln1b   = (const float*)d_in[12];
  const float* w1     = (const float*)d_in[13];
  const float* b1     = (const float*)d_in[14];
  const float* w2     = (const float*)d_in[15];
  const float* b2     = (const float*)d_in[16];
  const float* ln2g   = (const float*)d_in[17];
  const float* ln2b   = (const float*)d_in[18];
  const float* lnfg   = (const float*)d_in[19];
  const float* lnfb   = (const float*)d_in[20];
  const float* lmhead = (const float*)d_in[21];
  float* logits = (float*)d_out;

  char* base = (char*)d_ws;
  size_t off = 0;
  auto carve = [&](size_t bytes) -> char* {
    char* p = base + off;
    off += (bytes + 255) & ~(size_t)255;
    return p;
  };
  float*  h_f    = (float*) carve((size_t)Mtok * Dd * 4);
  __bf16* h_b    = (__bf16*)carve((size_t)Mtok * Dd * 2);
  __bf16* q_b    = (__bf16*)carve((size_t)Mtok * Dd * 2);
  __bf16* k_b    = (__bf16*)carve((size_t)Mtok * Dd * 2);
  __bf16* v_b    = (__bf16*)carve((size_t)Mtok * Dd * 2);
  __bf16* ctx_b  = (__bf16*)carve((size_t)Mtok * Dd * 2);
  float*  attn_f = (float*) carve((size_t)Mtok * Dd * 4);
  __bf16* ffn1_b = (__bf16*)carve((size_t)Mtok * Ff * 2);
  float*  ffn2_f = (float*) carve((size_t)Mtok * Dd * 4);
  __bf16* wq_b   = (__bf16*)carve((size_t)Ll * Dd * Dd * 2);
  __bf16* wk_b   = (__bf16*)carve((size_t)Ll * Dd * Dd * 2);
  __bf16* wv_b   = (__bf16*)carve((size_t)Ll * Dd * Dd * 2);
  __bf16* wo_b   = (__bf16*)carve((size_t)Ll * Dd * Dd * 2);
  __bf16* w1_b   = (__bf16*)carve((size_t)Ll * Dd * Ff * 2);
  __bf16* w2_b   = (__bf16*)carve((size_t)Ll * Ff * Dd * 2);
  __bf16* lmh_b  = (__bf16*)carve((size_t)Dd * Vv * 2);

  auto cvt = [&](const float* src, __bf16* dst, size_t n) {
    cvt_bf16_kernel<<<dim3((unsigned)((n + 255) / 256)), 256, 0, stream>>>(src, dst, n);
  };
  cvt(wq, wq_b, (size_t)Ll * Dd * Dd);
  cvt(wk, wk_b, (size_t)Ll * Dd * Dd);
  cvt(wv, wv_b, (size_t)Ll * Dd * Dd);
  cvt(wo, wo_b, (size_t)Ll * Dd * Dd);
  cvt(w1, w1_b, (size_t)Ll * Dd * Ff);
  cvt(w2, w2_b, (size_t)Ll * Ff * Dd);
  cvt(lmhead, lmh_b, (size_t)Dd * Vv);

  embed_kernel<<<dim3(Mtok, Dd / 256), 256, 0, stream>>>(ids, embw, pe, h_f, h_b);

  const dim3 gD (Dd / 128,  Mtok / 128);
  const dim3 gF (Ff / 128,  Mtok / 128);
  const dim3 gV (Vv / 128,  Mtok / 128);
  const dim3 gAtt(Ss / 64, Bb * Hh);

  for (int i = 0; i < Ll; ++i) {
    const size_t wDD = (size_t)i * Dd * Dd;
    const size_t wDF = (size_t)i * Dd * Ff;
    gemm_bf16_kernel<true, false, false, true><<<gD, 256, 0, stream>>>(
        h_b, wq_b + wDD, bq + (size_t)i * Dd, nullptr, q_b, Mtok, Dd, Dd);
    gemm_bf16_kernel<true, false, false, true><<<gD, 256, 0, stream>>>(
        h_b, wk_b + wDD, bk + (size_t)i * Dd, nullptr, k_b, Mtok, Dd, Dd);
    gemm_bf16_kernel<true, false, false, true><<<gD, 256, 0, stream>>>(
        h_b, wv_b + wDD, bv + (size_t)i * Dd, nullptr, v_b, Mtok, Dd, Dd);
    attention_kernel<<<gAtt, 128, 0, stream>>>(q_b, k_b, v_b, ctx_b);
    gemm_bf16_kernel<true, false, true, false><<<gD, 256, 0, stream>>>(
        ctx_b, wo_b + wDD, bo + (size_t)i * Dd, attn_f, nullptr, Mtok, Dd, Dd);
    layernorm_kernel<true><<<Mtok, 256, 0, stream>>>(
        h_f, attn_f, ln1g + (size_t)i * Dd, ln1b + (size_t)i * Dd, h_f, h_b);
    gemm_bf16_kernel<true, true, false, true><<<gF, 256, 0, stream>>>(
        h_b, w1_b + wDF, b1 + (size_t)i * Ff, nullptr, ffn1_b, Mtok, Ff, Dd);
    gemm_bf16_kernel<true, false, true, false><<<gD, 256, 0, stream>>>(
        ffn1_b, w2_b + wDF, b2 + (size_t)i * Dd, ffn2_f, nullptr, Mtok, Dd, Ff);
    layernorm_kernel<true><<<Mtok, 256, 0, stream>>>(
        h_f, ffn2_f, ln2g + (size_t)i * Dd, ln2b + (size_t)i * Dd, h_f, h_b);
  }
  layernorm_kernel<false><<<Mtok, 256, 0, stream>>>(
      h_f, nullptr, lnfg, lnfb, h_f, h_b);
  gemm_bf16_kernel<false, false, true, false><<<gV, 256, 0, stream>>>(
      h_b, lmh_b, nullptr, logits, nullptr, Mtok, Vv, Dd);
}
